// CharEmbedBLSTM_59725815218203
// MI455X (gfx1250) — matrix operand order repaired and checked by reading, not compiled, not run on hardware
//
#include <hip/hip_runtime.h>
#include <hip/hip_bf16.h>
#include <stdint.h>

#define VOCAB 128
#define EDIM  512
#define HDIM  256
#define G4H   1024          // 4*H
#define LL    16
#define NROWS 4096          // B*T
#define MTILE 32            // batch rows per block
#define VT    8             // vocab rows per precompute block

typedef __attribute__((ext_vector_type(16))) __bf16 v16bf;
typedef __attribute__((ext_vector_type(8)))  __bf16 v8bf;
typedef __attribute__((ext_vector_type(8)))  float  v8f;

__device__ __forceinline__ unsigned short f2bf(float f) {
    union { float f; unsigned u; } x; x.f = f;
    unsigned r = x.u + 0x7FFFu + ((x.u >> 16) & 1u);   // round-nearest-even
    return (unsigned short)(r >> 16);
}
__device__ __forceinline__ float bf2f(unsigned short b) {
    union { unsigned u; float f; } x; x.u = ((unsigned)b) << 16;
    return x.f;
}
__device__ __forceinline__ float sigmoidf_(float x) {
    return 1.0f / (1.0f + __expf(-x));
}
__device__ __forceinline__ float tanhf_(float x) {
    x = fminf(15.0f, fmaxf(-15.0f, x));
    float e = __expf(2.0f * x);
    return (e - 1.0f) / (e + 1.0f);
}

// ---------------------------------------------------------------------------
// pe[dir][v][c] = embed[v,:] . w_ih[orig(c),:] + b_ih[orig(c)] + b_hh[orig(c)]
// interleaved gate layout: c = 4*j + gate, orig(c) = gate*H + j
// Block = 8 vocab rows (LDS-cached) x all 1024 cols -> W_ih row reused 8x.
// ---------------------------------------------------------------------------
__global__ void precompute_pe(const float* __restrict__ embed,
                              const float* __restrict__ w_ih_f,
                              const float* __restrict__ b_ih_f,
                              const float* __restrict__ b_hh_f,
                              const float* __restrict__ w_ih_b,
                              const float* __restrict__ b_ih_b,
                              const float* __restrict__ b_hh_b,
                              float* __restrict__ pe) {
    __shared__ alignas(16) float emb[VT][EDIM];   // 16 KB
    const int v0 = blockIdx.x * VT, dir = blockIdx.y;
    const float* w  = dir ? w_ih_b : w_ih_f;
    const float* bi = dir ? b_ih_b : b_ih_f;
    const float* bh = dir ? b_hh_b : b_hh_f;
    for (int i = threadIdx.x; i < VT * EDIM; i += blockDim.x)
        ((float*)emb)[i] = embed[(size_t)v0 * EDIM + i];
    __syncthreads();
    for (int c = threadIdx.x; c < G4H; c += blockDim.x) {
        const int orig = (c & 3) * HDIM + (c >> 2);
        const float* wr = w + (size_t)orig * EDIM;
        float acc[VT];
        #pragma unroll
        for (int v = 0; v < VT; ++v) acc[v] = 0.0f;
        for (int e = 0; e < EDIM; e += 4) {
            float4 w4 = *(const float4*)&wr[e];
            #pragma unroll
            for (int v = 0; v < VT; ++v) {
                float4 a = *(const float4*)&emb[v][e];
                acc[v] = fmaf(a.x, w4.x, acc[v]);
                acc[v] = fmaf(a.y, w4.y, acc[v]);
                acc[v] = fmaf(a.z, w4.z, acc[v]);
                acc[v] = fmaf(a.w, w4.w, acc[v]);
            }
        }
        const float bias = bi[orig] + bh[orig];
        #pragma unroll
        for (int v = 0; v < VT; ++v)
            pe[((size_t)dir * VOCAB + v0 + v) * G4H + c] = acc[v] + bias;
    }
}

// ---------------------------------------------------------------------------
// Pack w_hh (f32 [4H,H]) into bf16 B-fragment layout:
//   bpack[dir][ki(8)][n_tile(64)][lane(32)][p(16)]
//   column (interleaved) = n_tile*16 + (lane&15); K = 32*ki + 8*(lane>>4) + (p&7) + 16*(p>>3)
// ---------------------------------------------------------------------------
__global__ void pack_whh(const float* __restrict__ w_hh_f,
                         const float* __restrict__ w_hh_b,
                         unsigned short* __restrict__ bpack) {
    const int idx  = blockIdx.x * blockDim.x + threadIdx.x;  // 524288 total
    const int p    = idx & 15;
    const int lane = (idx >> 4) & 31;
    const int nt   = (idx >> 9) & 63;
    const int ki   = (idx >> 15) & 7;
    const int dir  = (idx >> 18) & 1;
    const int c    = nt * 16 + (lane & 15);
    const int orig = (c & 3) * HDIM + (c >> 2);
    const int k    = 32 * ki + 8 * (lane >> 4) + (p & 7) + 16 * (p >> 3);
    const float* w = dir ? w_hh_b : w_hh_f;
    bpack[idx] = f2bf(w[(size_t)orig * HDIM + k]);
}

// ---------------------------------------------------------------------------
// Fused bidirectional LSTM recurrence. Block = 32 batch rows x one direction.
// 8 waves; per step each wave: 8 k-iters x (2 M-tiles x 8 N-tiles) WMMAs.
// h kept in LDS bf16 row-major [row][unit]; W_hh streamed from L2 each step
// with 8 B-fragments in flight per k-iter.
// ---------------------------------------------------------------------------
__global__ void __launch_bounds__(256)
blstm_kernel(const int* __restrict__ x,
             const float* __restrict__ pe,
             const unsigned short* __restrict__ bpack,
             float* __restrict__ out) {
    __shared__ alignas(16) unsigned short hbuf[MTILE][264];  // h, bf16 (rows 16B-aligned)
    __shared__ alignas(16) unsigned short gates[16][1028];   // one M-half of gate preacts
    __shared__ int xlds[MTILE][LL];                          // this block's char slice

    const int tid   = threadIdx.x;
    const int lane  = tid & 31;
    const int wave  = tid >> 5;
    const int dir   = blockIdx.y;
    const int r0    = blockIdx.x * MTILE;
    const int m     = lane & 15;
    const int khalf = lane >> 4;

    for (int i = tid; i < MTILE * 264; i += 256) ((unsigned short*)hbuf)[i] = 0;
    for (int i = tid; i < MTILE * LL; i += 256)
        ((int*)xlds)[i] = x[(size_t)(r0 + (i >> 4)) * LL + (i & 15)];

    float creg[MTILE];
    #pragma unroll
    for (int k = 0; k < MTILE; ++k) creg[k] = 0.0f;

    const unsigned short* bdir = bpack + (size_t)dir * (8 * 64 * 512);
    unsigned bofs = 0;   // stays 0; made opaque per step to defeat LICM/spilling

    #pragma unroll 1
    for (int t = 0; t < LL; ++t) {
        const int l = dir ? (LL - 1 - t) : t;
        __syncthreads();

        // opaque dependency: keeps W_hh loads inside the time loop (L2 stream)
        asm volatile("" : "+v"(bofs));
        const unsigned short* bt = bdir + bofs;

        // ---- gates = h_prev @ W_hh^T  (bf16 WMMA, f32 accum), 2 M-tiles ----
        v8f acc0[8], acc1[8];
        #pragma unroll
        for (int i = 0; i < 8; ++i) {
            acc0[i] = (v8f){0.f, 0.f, 0.f, 0.f, 0.f, 0.f, 0.f, 0.f};
            acc1[i] = (v8f){0.f, 0.f, 0.f, 0.f, 0.f, 0.f, 0.f, 0.f};
        }

        #pragma unroll
        for (int ki = 0; ki < 8; ++ki) {
            const int k0 = 32 * ki + 8 * khalf;
            // A fragments (16x32 bf16) for rows [0,16) and [16,32)
            v8bf a0lo = *(const v8bf*)&hbuf[m][k0];
            v8bf a0hi = *(const v8bf*)&hbuf[m][k0 + 16];
            v8bf a1lo = *(const v8bf*)&hbuf[m + 16][k0];
            v8bf a1hi = *(const v8bf*)&hbuf[m + 16][k0 + 16];
            v16bf a0 = __builtin_shufflevector(a0lo, a0hi,
                0, 1, 2, 3, 4, 5, 6, 7, 8, 9, 10, 11, 12, 13, 14, 15);
            v16bf a1 = __builtin_shufflevector(a1lo, a1hi,
                0, 1, 2, 3, 4, 5, 6, 7, 8, 9, 10, 11, 12, 13, 14, 15);

            // 8 B-fragments in flight (in-order returns -> per-WMMA waits)
            const unsigned short* bbase =
                bt + (((size_t)ki * 64 + wave * 8) * 32 + lane) * 16;
            v16bf bf[8];
            #pragma unroll
            for (int i = 0; i < 8; ++i)
                bf[i] = *(const v16bf*)(bbase + (size_t)i * 512);

            #pragma unroll
            for (int i = 0; i < 8; ++i) {
                acc0[i] = __builtin_amdgcn_wmma_f32_16x16x32_bf16(
                    false, a0, false, bf[i], (short)0, acc0[i], false, false);
                acc1[i] = __builtin_amdgcn_wmma_f32_16x16x32_bf16(
                    false, a1, false, bf[i], (short)0, acc1[i], false, false);
            }
        }

        // ---- two half-passes: spill one M-tile of gates, run the LSTM cell ----
        #pragma unroll
        for (int half = 0; half < 2; ++half) {
            #pragma unroll
            for (int i = 0; i < 8; ++i) {
                const int ncol = (wave * 8 + i) * 16 + (lane & 15);
                #pragma unroll
                for (int r = 0; r < 8; ++r) {
                    const float av = half ? acc1[i][r] : acc0[i][r];
                    gates[r + 8 * khalf][ncol] = f2bf(av);
                }
            }
            __syncthreads();

            // thread owns hidden unit j = tid for 16 rows of this half
            const int j = tid;
            #pragma unroll
            for (int kk = 0; kk < 16; ++kk) {
                const int k  = half * 16 + kk;
                const int n  = r0 + k;
                const int ch = xlds[k][l];
                const float4 pev =
                    *(const float4*)&pe[(((size_t)dir * VOCAB + ch) * G4H) + 4 * j];
                const unsigned short* gr = &gates[kk][4 * j];
                float gi = bf2f(gr[0]) + pev.x;
                float gf = bf2f(gr[1]) + pev.y;
                float gg = bf2f(gr[2]) + pev.z;
                float go = bf2f(gr[3]) + pev.w;
                float iv = sigmoidf_(gi);
                float fv = sigmoidf_(gf);
                float gv = tanhf_(gg);
                float ov = sigmoidf_(go);
                creg[k] = fv * creg[k] + iv * gv;
                float hv = ov * tanhf_(creg[k]);
                hbuf[k][j] = f2bf(hv);
                if (t == LL - 1)
                    out[(size_t)n * (2 * HDIM) + dir * HDIM + j] = hv;
            }
            __syncthreads();
        }
    }
}

// ---------------------------------------------------------------------------
extern "C" void kernel_launch(void* const* d_in, const int* in_sizes, int n_in,
                              void* d_out, int out_size, void* d_ws, size_t ws_size,
                              hipStream_t stream) {
    const int*   x       = (const int*)d_in[0];
    const float* embed   = (const float*)d_in[1];
    const float* w_ih_f  = (const float*)d_in[2];
    const float* w_hh_f  = (const float*)d_in[3];
    const float* b_ih_f  = (const float*)d_in[4];
    const float* b_hh_f  = (const float*)d_in[5];
    const float* w_ih_b  = (const float*)d_in[6];
    const float* w_hh_b  = (const float*)d_in[7];
    const float* b_ih_b  = (const float*)d_in[8];
    const float* b_hh_b  = (const float*)d_in[9];
    float* out = (float*)d_out;

    float* pe = (float*)d_ws;                                    // 2*128*1024 f32 = 1 MB
    unsigned short* bpack =
        (unsigned short*)((char*)d_ws + (size_t)2 * VOCAB * G4H * sizeof(float)); // 1 MB bf16

    precompute_pe<<<dim3(VOCAB / VT, 2), 256, 0, stream>>>(
        embed, w_ih_f, b_ih_f, b_hh_f, w_ih_b, b_ih_b, b_hh_b, pe);
    pack_whh<<<dim3(2048), 256, 0, stream>>>(w_hh_f, w_hh_b, bpack);
    blstm_kernel<<<dim3(NROWS / MTILE, 2), 256, 0, stream>>>(x, pe, bpack, out);
}
